// NormAdd_4561255268895
// MI455X (gfx1250) — matrix operand order repaired and checked by reading, not compiled, not run on hardware
//
#include <hip/hip_runtime.h>

// ---------------------------------------------------------------------------
// Fused: out = relu( concat(x0+b0, x1+b1) @ conv_w + conv_b )
// GEMM view: M = B*H*W = 131072, K = 512, N = 256, fp32 in/out.
// MI455X strategy:
//   * fold per-input biases + conv bias into beff[N] once per workgroup
//   * stage transposed weight as fp16 in LDS (260 KB -- uses CDNA5 320 KB WGP LDS)
//   * stream A (x0|x1) from HBM, double-buffering the next K-chunk in regs
//   * depth-2 rotating register buffer for B fragments from LDS so no WMMA
//     waits at s_wait_dscnt 0x0 (pipeline crosses K-chunk boundaries)
//   * v_wmma_f32_16x16x32_f16, f32 accumulators, ReLU + coalesced f32 stores
// Roofline: 268 MB read + 134 MB write @ 23.3 TB/s ~= 17 us floor; f16 WMMA
// compute (34.4 GFLOP) finishes well under that; fp32 WMMA (16x16x4) would
// be compute-bound -> fp16 is the right precision for the matrix path.
// ---------------------------------------------------------------------------

typedef __attribute__((ext_vector_type(16))) _Float16 v16h;
typedef __attribute__((ext_vector_type(8)))  float    v8f;
typedef __attribute__((ext_vector_type(4)))  float    v4f;

#define C_OUT     256
#define K_TOT     512
#define M_TOT     (8 * 128 * 128)          // 131072 pixels
#define WT_STRIDE 520                       // padded fp16 elems per column (bank spread)
#define NUM_WG    512
#define WAVES_WG  8
#define TILES_PER_WAVE ((M_TOT / 16) / (NUM_WG * WAVES_WG))   // == 2, exact

__global__ __launch_bounds__(256)
void conv1x1_concat_bias_relu_wmma(const float* __restrict__ x0,
                                   const float* __restrict__ x1,
                                   const float* __restrict__ b0,
                                   const float* __restrict__ b1,
                                   const float* __restrict__ Wf,   // [512,256] f32
                                   const float* __restrict__ cb,   // [256]
                                   float* __restrict__ out)        // [M,256] f32
{
    // 256*520*2 = 266,240 B + 1 KB bias: fits CDNA5's 320 KB workgroup LDS.
    __shared__ __align__(16) _Float16 sW[C_OUT * WT_STRIDE];  // sW[c][k] = Wf[k][c]
    __shared__ float sBias[C_OUT];

    const int tid = threadIdx.x;

    // ---- stage transposed fp16 weights into LDS (once per workgroup) ----
    for (int idx = tid; idx < K_TOT * C_OUT; idx += 256) {
        int k = idx >> 8;          // 0..511
        int c = idx & 255;         // 0..255
        sW[c * WT_STRIDE + k] = (_Float16)Wf[k * C_OUT + c];
    }
    // ---- effective bias: beff = b0@W_top + b1@W_bot + conv_b ----
    {
        float s = cb[tid];
        for (int k = 0; k < 256; ++k) {
            s += b0[k] * Wf[k * C_OUT + tid];
            s += b1[k] * Wf[(k + 256) * C_OUT + tid];
        }
        sBias[tid] = s;
    }
    __syncthreads();

    const int lane = tid & 31;
    const int wave = tid >> 5;
    const int row  = lane & 15;    // A: M row within tile / B & D: N column
    const int part = lane >> 4;    // half-wave selector

    // lane-invariant part of the B-fragment LDS address
    const _Float16* bbase = &sW[(size_t)row * WT_STRIDE + part * 16];
    // flattened fragment index i = kc*16 + nt  ->  LDS address
    auto bload = [&](int i) -> v16h {
        const int kc = i >> 4;
        const int nt = i & 15;
        return *(const v16h*)(bbase + (size_t)nt * 16 * WT_STRIDE + kc * 32);
    };

    for (int t = 0; t < TILES_PER_WAVE; ++t) {
        const int tile  = (blockIdx.x * WAVES_WG + wave) * TILES_PER_WAVE + t;
        const int mbase = tile * 16;

        // A-row pointers for this lane (lanes 0-15 and 16-31 share rows)
        const size_t prow = (size_t)(mbase + row) * C_OUT;
        const float* px0  = x0 + prow;
        const float* px1  = x1 + prow;
        const int    k0   = part * 16; // A layout: halves hold K {0-15}/{16-31} (matches B)

        // prefetch next tile's rows (global_prefetch_b8); uniform branch
        if (t + 1 < TILES_PER_WAVE) {
            const size_t pnext = (size_t)(mbase + 16 + row) * C_OUT + (size_t)part * 128;
            __builtin_prefetch(x0 + pnext, 0, 1);
            __builtin_prefetch(x1 + pnext, 0, 1);
        }

        // accumulators pre-loaded with effective bias (broadcast per N column)
        v8f acc[16];
#pragma unroll
        for (int nt = 0; nt < 16; ++nt) {
            float bv = sBias[nt * 16 + row];
#pragma unroll
            for (int r = 0; r < 8; ++r) acc[nt][r] = bv;
        }

        // ---- software pipelines ----
        // A chunk double buffer (global): chunk kc<8 -> x0, kc>=8 -> x1
        v4f c0, c1, c2, c3;
        {
            const float* s = px0;      // kc = 0
            c0 = *(const v4f*)(s + k0);
            c1 = *(const v4f*)(s + k0 + 4);
            c2 = *(const v4f*)(s + k0 + 8);
            c3 = *(const v4f*)(s + k0 + 12);
        }
        // B fragment depth-2 rotating buffer (LDS), crosses kc boundaries
        v16h bcur = bload(0);
        v16h bnxt = bload(1);

#pragma unroll
        for (int kc = 0; kc < 16; ++kc) {
            // issue next A chunk's loads before consuming current chunk
            v4f n0, n1, n2, n3;
            if (kc + 1 < 16) {
                const float* s = (kc + 1 < 8) ? (px0 + (kc + 1) * 32)
                                              : (px1 + (kc + 1 - 8) * 32);
                n0 = *(const v4f*)(s + k0);
                n1 = *(const v4f*)(s + k0 + 4);
                n2 = *(const v4f*)(s + k0 + 8);
                n3 = *(const v4f*)(s + k0 + 12);
            }

            v16h a;
            a[0]  = (_Float16)c0.x;  a[1]  = (_Float16)c0.y;
            a[2]  = (_Float16)c0.z;  a[3]  = (_Float16)c0.w;
            a[4]  = (_Float16)c1.x;  a[5]  = (_Float16)c1.y;
            a[6]  = (_Float16)c1.z;  a[7]  = (_Float16)c1.w;
            a[8]  = (_Float16)c2.x;  a[9]  = (_Float16)c2.y;
            a[10] = (_Float16)c2.z;  a[11] = (_Float16)c2.w;
            a[12] = (_Float16)c3.x;  a[13] = (_Float16)c3.y;
            a[14] = (_Float16)c3.z;  a[15] = (_Float16)c3.w;

            // one A fragment feeds all 16 N-tiles; B stays 2 fragments ahead
#pragma unroll
            for (int nt = 0; nt < 16; ++nt) {
                const int i = kc * 16 + nt;        // compile-time constant
                v16h bnn = (i + 2 < 256) ? bload(i + 2) : bcur;
                acc[nt] = __builtin_amdgcn_wmma_f32_16x16x32_f16(
                    false, a, false, bcur, (short)0, acc[nt], false, false);
                bcur = bnxt;
                bnxt = bnn;
            }

            c0 = n0; c1 = n1; c2 = n2; c3 = n3;
        }

        // ---- ReLU + store (C/D layout: vgpr r -> M = part*8 + r, lane&15 -> N) ----
#pragma unroll
        for (int nt = 0; nt < 16; ++nt) {
            const int n = nt * 16 + row;
#pragma unroll
            for (int r = 0; r < 8; ++r) {
                const int m = mbase + part * 8 + r;
                float v = acc[nt][r];
                out[(size_t)m * C_OUT + n] = v > 0.0f ? v : 0.0f;
            }
        }
    }
}

extern "C" void kernel_launch(void* const* d_in, const int* in_sizes, int n_in,
                              void* d_out, int out_size, void* d_ws, size_t ws_size,
                              hipStream_t stream) {
    (void)in_sizes; (void)n_in; (void)d_ws; (void)ws_size; (void)out_size;
    const float* x0 = (const float*)d_in[0];
    const float* x1 = (const float*)d_in[1];
    const float* b0 = (const float*)d_in[2];
    const float* b1 = (const float*)d_in[3];
    const float* W  = (const float*)d_in[4];
    const float* cb = (const float*)d_in[5];
    float* out = (float*)d_out;

    hipLaunchKernelGGL(conv1x1_concat_bias_relu_wmma,
                       dim3(NUM_WG), dim3(256), 0, stream,
                       x0, x1, b0, b1, W, cb, out);
}
